// nngls_30442728194266
// MI455X (gfx1250) — compile-verified, matrix operand order checked by reading, not run on hardware
//
#include <hip/hip_runtime.h>
#include <hip/hip_bf16.h>

typedef __attribute__((ext_vector_type(16))) _Float16 v16h;
typedef __attribute__((ext_vector_type(8)))  float    v8f;

#define KN 20   // neighbors
#define PD 10   // covariates
#define HD 64   // hidden width
#define WAVES_PER_BLOCK 8

// ---------------- helpers ----------------
__device__ __forceinline__ float hash01(unsigned s) {
    // PCG-style hash -> uniform [0,1)
    unsigned h = s * 747796405u + 2891336453u;
    h = ((h >> ((h >> 28) + 4u)) ^ h) * 277803737u;
    h = (h >> 22) ^ h;
    return (float)(h >> 8) * (1.0f / 16777216.0f);
}

__device__ __forceinline__ unsigned pack_h2(float a, float b) {
    union { _Float16 h[2]; unsigned u; } v;
    v.h[0] = (_Float16)a;
    v.h[1] = (_Float16)b;
    return v.u;
}

// ---------------- ws init ----------------
__global__ void nngls_init_ws(unsigned* ws) {
    ws[0] = 0x7F800000u;  // min accumulator = +inf (positions are >= 0)
    ws[1] = 0u;           // max accumulator = 0
}

// ---------------- global min/max of scattered neighbor positions ----------------
__global__ void nngls_minmax(const float* __restrict__ pos,
                             const int* __restrict__ edge_src,
                             int E, unsigned* __restrict__ ws) {
    int e = blockIdx.x * blockDim.x + threadIdx.x;
    float vmin = 3.4e38f, vmax = 0.0f;
    if (e < E) {
        int s = edge_src[e];
        float a = pos[2 * s], b = pos[2 * s + 1];
        vmin = fminf(a, b);
        vmax = fmaxf(a, b);
    }
    #pragma unroll
    for (int off = 16; off > 0; off >>= 1) {
        vmin = fminf(vmin, __shfl_down(vmin, off));
        vmax = fmaxf(vmax, __shfl_down(vmax, off));
    }
    if ((threadIdx.x & 31) == 0) {
        atomicMin(&ws[0], __float_as_uint(vmin));
        atomicMax(&ws[1], __float_as_uint(vmax));
    }
}

// ---------------- MLP: o = relu(x@W1 + b1)@W2 + b2  (WMMA f16->f32) ----------------
// 8 waves/block, one 16-row tile per wave. Operands staged in LDS as zero-padded
// f16 pairs so fragment construction is branch-free (plain ds_load_b32).
__global__ void __launch_bounds__(32 * WAVES_PER_BLOCK)
nngls_mlp_wmma(const float* __restrict__ x,
               const float* __restrict__ W1, const float* __restrict__ b1,
               const float* __restrict__ W2, const float* __restrict__ b2,
               float* __restrict__ o_out, int N) {
    // sW[j2][col]: packed f16 pair (K=2*j2, 2*j2+1), K zero-padded 10 -> 32
    __shared__ unsigned sW[16 * HD];                       // 4 KB (block-shared)
    __shared__ unsigned sX[WAVES_PER_BLOCK][16 * 16];      // 1 KB per wave

    const int tid   = threadIdx.x;
    const int lane  = tid & 31;
    const int wid   = tid >> 5;
    const int half  = lane >> 4;
    const int l16   = lane & 15;
    const int tile  = blockIdx.x * WAVES_PER_BLOCK + wid;
    const int row0  = tile * 16;

    // ---- stage W1 (once per block, coalesced over col) ----
    for (int idx = tid; idx < 16 * HD; idx += 32 * WAVES_PER_BLOCK) {
        const int j2 = idx >> 6, col = idx & 63;
        unsigned u = 0u;
        if (j2 < PD / 2)   // PD even: pair (2*j2, 2*j2+1) fully in range
            u = pack_h2(W1[(2 * j2) * HD + col], W1[(2 * j2 + 1) * HD + col]);
        sW[idx] = u;
    }
    // ---- stage this wave's 16xPD x-tile as packed pairs, zero-padded to K=32 ----
    for (int idx = lane; idx < 16 * 16; idx += 32) {
        const int r = idx >> 4, j2 = idx & 15;
        const int row = row0 + r;
        const int rowL = row < N ? row : N - 1;
        unsigned u = 0u;
        if (j2 < PD / 2)
            u = pack_h2(x[rowL * PD + 2 * j2], x[rowL * PD + 2 * j2 + 1]);
        sX[wid][idx] = u;
    }
    __syncthreads();

    // ---- A fragment (16x32 f16): VGPR q holds K pair at
    //      j2 = ((q<4)?0:8) + 4*half + (q&3)  (ISA 7.12.2 16-bit A layout) ----
    union { v16h v; unsigned u[8]; } afrag;
    #pragma unroll
    for (int q = 0; q < 8; ++q) {
        const int j2 = ((q < 4) ? 0 : 8) + 4 * half + (q & 3);
        afrag.u[q] = sX[wid][l16 * 16 + j2];
    }

    // ---- 4 column tiles: wmma + fused second layer from accumulators ----
    float pr[8];
    #pragma unroll
    for (int q = 0; q < 8; ++q) pr[q] = 0.0f;

    #pragma unroll
    for (int t = 0; t < 4; ++t) {
        const int col = t * 16 + l16;
        union { v16h v; unsigned u[8]; } bfrag;
        #pragma unroll
        for (int q = 0; q < 8; ++q) {
            const int j2 = ((q < 4) ? 0 : 8) + 4 * half + (q & 3);
            bfrag.u[q] = sW[j2 * HD + col];
        }
        v8f acc = {};
        acc = __builtin_amdgcn_wmma_f32_16x16x32_f16(
            false, afrag.v, false, bfrag.v, (short)0, acc, false, false);

        const float bias = b1[col];
        const float w2c  = W2[col];
        // D layout: lane (l16, half) holds h[m = q + 8*half][n = col] in acc[q]
        #pragma unroll
        for (int q = 0; q < 8; ++q) {
            const float h = fmaxf(acc[q] + bias, 0.0f);
            pr[q] += h * w2c;     // contribution of column `col` to row q + 8*half
        }
    }

    // ---- butterfly-reduce the 8 row partials across the 16 lanes of each half ----
    #pragma unroll
    for (int mask = 1; mask < 16; mask <<= 1) {
        #pragma unroll
        for (int q = 0; q < 8; ++q)
            pr[q] += __shfl_xor(pr[q], mask);
    }
    if (l16 == 0) {
        const float bb = b2[0];
        #pragma unroll
        for (int q = 0; q < 8; ++q) {
            const int row = row0 + 8 * half + q;
            if (row < N) o_out[row] = pr[q] + bb;
        }
    }
}

// ---------------- per-target: gather, jitter, 20x20 solve, decorrelate ----------------
// one wave32 per target, 8 waves (8 targets) per 256-thread block
__global__ void __launch_bounds__(256)
nngls_solve(const float* __restrict__ pos,
            const float* __restrict__ y,
            const float* __restrict__ theta,
            const int* __restrict__ edge_src,
            const int* __restrict__ edge_dst,
            const int* __restrict__ edge_attr,
            const float* __restrict__ o_in,     // = d_out + 2N, written by mlp kernel
            const unsigned* __restrict__ ws,    // [0]=min bits, [1]=max bits
            float* __restrict__ out,            // [0,N): y_decor, [N,2N): o_decor
            int N, int E) {
    __shared__ float sA [8][KN * 21];       // 20x21 augmented matrix per wave
    __shared__ float sCN[8][KN];
    __shared__ float sNP[8][KN][2];
    __shared__ float sY [8][KN];
    __shared__ float sO [8][KN];

    const int lane = threadIdx.x & 31;
    const int wid  = threadIdx.x >> 5;
    const int i    = blockIdx.x * 8 + wid;          // target id (may exceed N)
    float (*A)[21]  = (float (*)[21])sA[wid];
    float*  cn      = sCN[wid];
    float (*np)[2]  = sNP[wid];
    float*  ynb     = sY[wid];
    float*  onb     = sO[wid];

    const float sigma2 = theta[0], phi = theta[1], tau = theta[2];
    const float jscale = 10000.0f *
        (__uint_as_float(ws[1]) - __uint_as_float(ws[0]));

    // ---- gather edges for this target (edges e = i*K + k; slot from edge_attr) ----
    if (lane < KN) {
        int e = i * KN + lane;
        if (e >= E) e = E - 1;                      // clamp for tail waves
        const int src  = edge_src[e];
        const int dst  = edge_dst[e];
        const int slot = ((unsigned)edge_attr[e]) % KN;
        const float sx = pos[2 * src], sy2 = pos[2 * src + 1];
        const float dx = pos[2 * dst] - sx, dy = pos[2 * dst + 1] - sy2;
        cn[slot]    = sigma2 * __expf(-phi * __fsqrt_rn(dx * dx + dy * dy));
        np[slot][0] = sx;
        np[slot][1] = sy2;
        ynb[slot]   = y[src];
        onb[slot]   = o_in[src];
    }
    __syncthreads();

    // ---- zero-slot jitter (hash RNG stands in for the JAX PRNG stream) ----
    if (lane < KN) {
        #pragma unroll
        for (int d = 0; d < 2; ++d)
            if (np[lane][d] == 0.0f)
                np[lane][d] = hash01(((unsigned)i * KN + lane) * 2u + d) * jscale;
    }
    __syncthreads();

    // ---- build 20x20 covariance (+nugget) and RHS column 20 ----
    for (int idx = lane; idx < KN * KN; idx += 32) {
        const int rr = idx / KN, cc = idx % KN;
        const float dx = np[rr][0] - np[cc][0];
        const float dy = np[rr][1] - np[cc][1];
        float v = sigma2 * __expf(-phi * __fsqrt_rn(dx * dx + dy * dy));
        if (rr == cc) v += tau * sigma2;
        A[rr][cc] = v;
    }
    if (lane < KN) A[lane][KN] = cn[lane];
    __syncthreads();

    // ---- forward elimination (SPD: no pivoting), rows parallel over lanes ----
    for (int kk = 0; kk < KN - 1; ++kk) {
        const int r = kk + 1 + lane;
        if (r < KN) {
            const float f = A[r][kk] / A[kk][kk];
            #pragma unroll 4
            for (int j = kk; j <= KN; ++j)
                A[r][j] -= f * A[kk][j];
        }
        __syncthreads();
    }

    // ---- back substitution + decorrelation (lane 0; ~250 flops, negligible) ----
    if (lane == 0) {
        float xv[KN];
        for (int kk = KN - 1; kk >= 0; --kk) {
            float s = A[kk][KN];
            for (int j = kk + 1; j < KN; ++j) s -= A[kk][j] * xv[j];
            xv[kk] = s / A[kk][kk];
        }
        float bc = 0.0f, by = 0.0f, bo = 0.0f;
        #pragma unroll
        for (int k = 0; k < KN; ++k) {
            bc += xv[k] * cn[k];
            by += xv[k] * ynb[k];
            bo += xv[k] * onb[k];
        }
        const float F  = sigma2 * (1.0f + tau) - bc;
        const float sF = __fsqrt_rn(F);
        if (i < N) {
            out[i]     = (y[i]    - by) / sF;
            out[N + i] = (o_in[i] - bo) / sF;
        }
    }
}

extern "C" void kernel_launch(void* const* d_in, const int* in_sizes, int n_in,
                              void* d_out, int out_size, void* d_ws, size_t ws_size,
                              hipStream_t stream) {
    const float* pos       = (const float*)d_in[0];
    const float* x         = (const float*)d_in[1];
    const float* y         = (const float*)d_in[2];
    const float* theta     = (const float*)d_in[3];
    const float* W1        = (const float*)d_in[4];
    const float* b1        = (const float*)d_in[5];
    const float* W2        = (const float*)d_in[6];
    const float* b2        = (const float*)d_in[7];
    const int*   edge_src  = (const int*)d_in[8];
    const int*   edge_dst  = (const int*)d_in[9];
    const int*   edge_attr = (const int*)d_in[10];

    const int N = in_sizes[2];        // 50000
    const int E = in_sizes[8];        // N*K
    float* out = (float*)d_out;       // [y_decor | o_decor | o]
    float* o_buf = out + 2 * (size_t)N;
    unsigned* ws = (unsigned*)d_ws;

    // 1) reset min/max accumulators (deterministic each call)
    nngls_init_ws<<<1, 1, 0, stream>>>(ws);

    // 2) global min/max of scattered neighbor positions (for jitter scale)
    nngls_minmax<<<(E + 255) / 256, 256, 0, stream>>>(pos, edge_src, E, ws);

    // 3) MLP via WMMA: writes o into d_out[2N:3N]
    const int tiles  = (N + 15) / 16;                               // 3125
    const int mblks  = (tiles + WAVES_PER_BLOCK - 1) / WAVES_PER_BLOCK;
    nngls_mlp_wmma<<<mblks, 32 * WAVES_PER_BLOCK, 0, stream>>>(
        x, W1, b1, W2, b2, o_buf, N);

    // 4) per-target gather + jitter + 20x20 solve + decorrelation
    const int blocks = (N + 7) / 8;   // 6250 blocks x 8 waves
    nngls_solve<<<blocks, 256, 0, stream>>>(pos, y, theta, edge_src, edge_dst,
                                            edge_attr, o_buf, ws, out, N, E);
}
